// TemporalGraphWaveNet_78606491451781
// MI455X (gfx1250) — matrix operand order repaired and checked by reading, not compiled, not run on hardware
//
#include <hip/hip_runtime.h>
#include <hip/hip_bf16.h>

typedef __attribute__((ext_vector_type(16))) _Float16 v16h;
typedef __attribute__((ext_vector_type(8)))  _Float16 v8h;
typedef __attribute__((ext_vector_type(8)))  float    v8f;

#define NN      100000
#define NE      1600000
#define TS      8
#define FIN     64

// ---------------------------------------------------------------------------
// Zero helper (grid-stride float4)
// ---------------------------------------------------------------------------
__global__ void zero_f32(float* __restrict__ p, long long n4) {
  long long i = (long long)blockIdx.x * blockDim.x + threadIdx.x;
  long long stride = (long long)gridDim.x * blockDim.x;
  float4* p4 = (float4*)p;
  float4 z = make_float4(0.f, 0.f, 0.f, 0.f);
  for (; i < n4; i += stride) p4[i] = z;
}

// ---------------------------------------------------------------------------
// Weight prep: convert f32 weights into f16, pre-swizzled into the WMMA
// B-matrix (32x16 f16) fragment layout. Flat index:
//   ((g)*O + o)*16 + i  with  g = kk*2 + halfsel,  element = W[K = kk*32 + halfsel*16 + i][o]
// so each lane's 16-half fragment is one contiguous 32-byte load.
// ---------------------------------------------------------------------------
__global__ void prep_conv_w(const float* __restrict__ w, _Float16* __restrict__ o, int O) {
  // w: [O][FIN][3] row-major; im2col K = tap*FIN + f, total K = 192
  int idx = blockIdx.x * blockDim.x + threadIdx.x;
  int total = 192 * O;
  if (idx >= total) return;
  int i  = idx & 15;
  int oo = (idx >> 4) % O;
  int g  = idx / (16 * O);                   // 0..11
  int K  = (g >> 1) * 32 + (g & 1) * 16 + i; // 0..191
  int tap = K / FIN, f = K % FIN;
  o[idx] = (_Float16)w[(oo * FIN + f) * 3 + tap];
}

__global__ void prep_temp_w(const float* __restrict__ w, _Float16* __restrict__ o,
                            int O2, int K) {
  // w: [O2][K] row-major, K index = och*TS + t
  int idx = blockIdx.x * blockDim.x + threadIdx.x;
  int total = K * O2;
  if (idx >= total) return;
  int i  = idx & 15;
  int oo = (idx >> 4) % O2;
  int g  = idx / (16 * O2);
  int k  = (g >> 1) * 32 + (g & 1) * 16 + i;
  o[idx] = (_Float16)w[oo * K + k];
}

// ---------------------------------------------------------------------------
// Edge scatter: agg[dst, t, f] += xin[src, f]   (one thread per edge-feature)
// ---------------------------------------------------------------------------
__global__ void scatter_kernel(const float* __restrict__ xin,
                               const int* __restrict__ src,
                               const int* __restrict__ dst,
                               const int* __restrict__ tt,
                               float* __restrict__ agg, int E) {
  long long idx = (long long)blockIdx.x * blockDim.x + threadIdx.x;
  if (idx >= (long long)E * FIN) return;
  int e = (int)(idx >> 6);
  int f = (int)(idx & 63);
  int s = src[e];
  int d = dst[e];
  int t = tt[e];
  atomicAdd(&agg[((long long)d * TS + t) * FIN + f], xin[(long long)s * FIN + f]);
}

// ---------------------------------------------------------------------------
// Fused layer: self-loop add + f16 cast -> LDS, conv-as-GEMM (K=192) via WMMA,
// ReLU+bias -> f16 h in LDS (o*T+t order), temporal GEMM (K=F_OUT*8) via WMMA,
// f32 output. One wave (32 threads) handles a 16-node tile.
// ---------------------------------------------------------------------------
template <int F_OUT>
__global__ __launch_bounds__(32) void layer_kernel(
    const float* __restrict__ agg,       // [N, TS, FIN]
    const float* __restrict__ xin,       // [N, FIN]
    const _Float16* __restrict__ wBconv, // swizzled [192 x F_OUT]
    const float* __restrict__ bconv,     // [F_OUT]
    const _Float16* __restrict__ wBtemp, // swizzled [(F_OUT*8) x F_OUT]
    const float* __restrict__ btemp,     // [F_OUT]
    float* __restrict__ out,             // [N, F_OUT]
    int n_nodes) {
  constexpr int NT = F_OUT / 16;           // N-tiles (4 or 2)
  constexpr int KH = F_OUT * TS;           // h features per node (512 or 256)
  constexpr int KSTEPS = KH / 32;          // temporal k-steps (16 or 8)

  __shared__ __align__(32) _Float16 s_agg[16 * (TS + 2) * FIN]; // padded T halo
  __shared__ __align__(32) _Float16 s_h[16 * KH];

  const int lane = threadIdx.x;            // 0..31 (wave32)
  const int node0 = blockIdx.x * 16;
  const int halfsel = (lane >= 16) ? 1 : 0;
  const int kb_lo = halfsel * 16;          // A-fragment low K base per lane half
  const int colN = lane & 15;              // B/C/D column per lane

  __builtin_prefetch(wBconv, 0, 0);
  __builtin_prefetch(wBtemp, 0, 0);

  // ---- stage agg + self-loop into LDS as f16, zero T-halo ----
  for (int i = lane; i < 16 * FIN; i += 32) {
    int n = i >> 6, f = i & 63;
    s_agg[(n * (TS + 2) + 0) * FIN + f] = (_Float16)0.f;
    s_agg[(n * (TS + 2) + (TS + 1)) * FIN + f] = (_Float16)0.f;
  }
  for (int c = lane; c < 16 * TS * (FIN / 4); c += 32) {
    int n = c >> 7;            // / (TS * 16)
    int rem = c & 127;
    int t = rem >> 4;
    int f4 = rem & 15;
    int node = node0 + n;
    float4 v = make_float4(0.f, 0.f, 0.f, 0.f);
    if (node < n_nodes) {
      const float4* ap = (const float4*)(agg + ((long long)node * TS + t) * FIN);
      const float4* xp = (const float4*)(xin + (long long)node * FIN);
      float4 a = ap[f4];
      float4 x = xp[f4];
      v = make_float4(a.x + x.x, a.y + x.y, a.z + x.z, a.w + x.w);
    }
    _Float16* dp = &s_agg[(n * (TS + 2) + t + 1) * FIN + f4 * 4];
    dp[0] = (_Float16)v.x; dp[1] = (_Float16)v.y;
    dp[2] = (_Float16)v.z; dp[3] = (_Float16)v.w;
  }
  __syncthreads();

  // ---- conv GEMM: rows = (node,t) pairs, K = 192 (3 taps x 64 feat) ----
  const int arow = lane & 15;              // A-matrix row per lane
  for (int mt = 0; mt < 8; ++mt) {
    int row = mt * 16 + arow;
    int nl = row >> 3, tloc = row & 7;
    v8f acc[NT];
#pragma unroll
    for (int ot = 0; ot < NT; ++ot) acc[ot] = (v8f){0.f,0.f,0.f,0.f,0.f,0.f,0.f,0.f};
#pragma unroll
    for (int kk = 0; kk < 6; ++kk) {
      int tap = kk >> 1;
      int fbase = (kk & 1) * 32;
      const _Float16* ab = &s_agg[((nl * (TS + 2)) + tloc + tap) * FIN + fbase + kb_lo];
      v8h lo = *(const v8h*)(ab);
      v8h hi = *(const v8h*)(ab + 8);
      v16h a = __builtin_shufflevector(lo, hi, 0,1,2,3,4,5,6,7,8,9,10,11,12,13,14,15);
#pragma unroll
      for (int ot = 0; ot < NT; ++ot) {
        const v16h b = *(const v16h*)(wBconv +
            ((size_t)((kk * 2 + halfsel) * F_OUT + ot * 16 + colN)) * 16);
        acc[ot] = __builtin_amdgcn_wmma_f32_16x16x32_f16(
            false, a, false, b, (short)0, acc[ot], false, false);
      }
    }
    // bias + ReLU -> f16 h in LDS at [node][o*TS + t]
#pragma unroll
    for (int ot = 0; ot < NT; ++ot) {
      float bias = bconv[ot * 16 + colN];
#pragma unroll
      for (int r = 0; r < 8; ++r) {
        int m = r + halfsel * 8;
        int row2 = mt * 16 + m;
        int nl2 = row2 >> 3, t2 = row2 & 7;
        float v = acc[ot][r] + bias;
        v = v > 0.f ? v : 0.f;
        s_h[nl2 * KH + (ot * 16 + colN) * TS + t2] = (_Float16)v;
      }
    }
  }
  __syncthreads();

  // ---- temporal GEMM: rows = nodes (16), K = F_OUT*8 ----
  {
    v8f acc[NT];
#pragma unroll
    for (int ot = 0; ot < NT; ++ot) acc[ot] = (v8f){0.f,0.f,0.f,0.f,0.f,0.f,0.f,0.f};
#pragma unroll
    for (int kk = 0; kk < KSTEPS; ++kk) {
      const _Float16* ab = &s_h[arow * KH + kk * 32 + kb_lo];
      v8h lo = *(const v8h*)(ab);
      v8h hi = *(const v8h*)(ab + 8);
      v16h a = __builtin_shufflevector(lo, hi, 0,1,2,3,4,5,6,7,8,9,10,11,12,13,14,15);
#pragma unroll
      for (int ot = 0; ot < NT; ++ot) {
        const v16h b = *(const v16h*)(wBtemp +
            ((size_t)((kk * 2 + halfsel) * F_OUT + ot * 16 + colN)) * 16);
        acc[ot] = __builtin_amdgcn_wmma_f32_16x16x32_f16(
            false, a, false, b, (short)0, acc[ot], false, false);
      }
    }
#pragma unroll
    for (int ot = 0; ot < NT; ++ot) {
      float bias = btemp[ot * 16 + colN];
#pragma unroll
      for (int r = 0; r < 8; ++r) {
        int node = node0 + r + halfsel * 8;
        if (node < n_nodes)
          out[(long long)node * F_OUT + ot * 16 + colN] = acc[ot][r] + bias;
      }
    }
  }
}

// ---------------------------------------------------------------------------
extern "C" void kernel_launch(void* const* d_in, const int* in_sizes, int n_in,
                              void* d_out, int out_size, void* d_ws, size_t ws_size,
                              hipStream_t stream) {
  (void)in_sizes; (void)n_in; (void)out_size; (void)ws_size;
  const float* x   = (const float*)d_in[0];
  const int*   ei  = (const int*)d_in[1];     // [2, E]
  const int*   ti  = (const int*)d_in[2];     // [E]
  const float* w1c = (const float*)d_in[3];
  const float* b1c = (const float*)d_in[4];
  const float* w1t = (const float*)d_in[5];
  const float* b1t = (const float*)d_in[6];
  const float* w2c = (const float*)d_in[7];
  const float* b2c = (const float*)d_in[8];
  const float* w2t = (const float*)d_in[9];
  const float* b2t = (const float*)d_in[10];
  float* out = (float*)d_out;

  char* ws = (char*)d_ws;
  size_t off = 0;
  auto carve = [&](size_t bytes) -> char* {
    char* p = ws + off;
    off = (off + bytes + 255) & ~(size_t)255;
    return p;
  };
  float*     agg1 = (float*)carve((size_t)NN * TS * FIN * 4);
  float*     agg2 = (float*)carve((size_t)NN * TS * FIN * 4);
  float*     h1   = (float*)carve((size_t)NN * 64 * 4);
  _Float16*  w1cs = (_Float16*)carve(192 * 64 * 2);
  _Float16*  w1ts = (_Float16*)carve(512 * 64 * 2);
  _Float16*  w2cs = (_Float16*)carve(192 * 32 * 2);
  _Float16*  w2ts = (_Float16*)carve(256 * 32 * 2);

  const int* src = ei;
  const int* dst = ei + NE;

  // zero agg accumulators (every call)
  long long n4 = (long long)NN * TS * FIN / 4;
  zero_f32<<<4096, 256, 0, stream>>>(agg1, n4);
  zero_f32<<<4096, 256, 0, stream>>>(agg2, n4);

  // weight prep (f16 swizzled fragments)
  prep_conv_w<<<(192 * 64 + 255) / 256, 256, 0, stream>>>(w1c, w1cs, 64);
  prep_temp_w<<<(512 * 64 + 255) / 256, 256, 0, stream>>>(w1t, w1ts, 64, 512);
  prep_conv_w<<<(192 * 32 + 255) / 256, 256, 0, stream>>>(w2c, w2cs, 32);
  prep_temp_w<<<(256 * 32 + 255) / 256, 256, 0, stream>>>(w2t, w2ts, 32, 256);

  long long scThreads = (long long)NE * FIN;
  int scBlocks = (int)((scThreads + 255) / 256);
  int lyBlocks = (NN + 15) / 16;

  // layer 1
  scatter_kernel<<<scBlocks, 256, 0, stream>>>(x, src, dst, ti, agg1, NE);
  layer_kernel<64><<<lyBlocks, 32, 0, stream>>>(agg1, x, w1cs, b1c, w1ts, b1t, h1, NN);

  // layer 2
  scatter_kernel<<<scBlocks, 256, 0, stream>>>(h1, src, dst, ti, agg2, NE);
  layer_kernel<32><<<lyBlocks, 32, 0, stream>>>(agg2, h1, w2cs, b2c, w2ts, b2t, out, NN);
}
